// Attention_28235115004180
// MI455X (gfx1250) — matrix operand order repaired and checked by reading, not compiled, not run on hardware
//
#include <hip/hip_runtime.h>

// Problem constants (B=1)
#define S_LEN 1024
#define DIM   4096
#define NH    32
#define NC    8
#define HD    128
#define QK_SCALE 0.08838834764831845f   // 1/sqrt(128)

typedef __attribute__((ext_vector_type(16))) __bf16 v16bf;
typedef __attribute__((ext_vector_type(8)))  __bf16 v8bf;
typedef __attribute__((ext_vector_type(4)))  __bf16 v4bf;
typedef __attribute__((ext_vector_type(8)))  float  v8f;
typedef __attribute__((ext_vector_type(4)))  int    v4i;

// Address-space-qualified pointee types for the CDNA5 async global->LDS
// builtin: (global v4i* src, lds v4i* dst, imm offset, imm cpol).
typedef __attribute__((address_space(1))) v4i gv4i;
typedef __attribute__((address_space(3))) v4i lv4i;

union BFrag { v16bf v; v8bf h[2]; };

// CDNA5 async global->LDS copy (ASYNCcnt) if the toolchain exposes it;
// otherwise classic global_load + ds_store staging (same pipeline shape).
#if __has_builtin(__builtin_amdgcn_global_load_async_to_lds_b128)
#define ASYNC_LDS 1
#else
#define ASYNC_LDS 0
#endif

__device__ __forceinline__ void stage_row32(const __bf16* __restrict__ g,
                                            __bf16* __restrict__ l)
{
#if ASYNC_LDS
#pragma unroll
  for (int i = 0; i < 4; ++i)
    __builtin_amdgcn_global_load_async_to_lds_b128(
        (gv4i*)(g + 8 * i), (lv4i*)(l + 8 * i), 0, 0);
#else
#pragma unroll
  for (int i = 0; i < 4; ++i)
    *(v8bf*)(l + 8 * i) = *(const v8bf*)(g + 8 * i);
#endif
}

__device__ __forceinline__ void stage_commit_wait()
{
#if ASYNC_LDS
#if __has_builtin(__builtin_amdgcn_s_wait_asynccnt)
  __builtin_amdgcn_s_wait_asynccnt(0);
#else
  asm volatile("s_wait_asynccnt 0x0" ::: "memory");
#endif
#endif
  __syncthreads();
}

// ---------------------------------------------------------------------------
// WMMA fragment layouts (CDNA5 ISA 7.12.2), bf16 16x16x32:
//   A 16x32 : lane&15 = M ; halves 0-7 -> K = 8*(lane>>4)+h,
//                           halves 8-15 -> K = 16 + 8*(lane>>4)+(h-8)
//   B 32x16 : lane&15 = N ; this code packs A and B alike: half h -> K = 16*(lane>>4) + h
//   C 16x16 : lane&15 = N ; vgpr r -> M = r + 8*(lane>>4)
// All operands stored K-contiguous -> contiguous 16B loads.
// ---------------------------------------------------------------------------

// ===========================================================================
// Big weight GEMM: C(f32, MxN) = A(MxK) @ B(NxK)^T, M,N multiples of 128,
// row stride of A and B == K. 128x128 block tile, 8 waves, 32x64 per wave.
// LDS double-buffered k-steps of 32, fed by async global->LDS copies.
// ===========================================================================
#define BK 32
#define PAD_ROW 40   // padded LDS row (elements): 80B keeps b128 reads aligned
                     // and spreads rows across banks (20-dword stride)

__global__ __launch_bounds__(256) void gemm_block_kernel(
    const __bf16* __restrict__ A,
    const __bf16* __restrict__ B,
    float* __restrict__ C, int ldc, int K)
{
  __shared__ __bf16 lds[2][256 * PAD_ROW];   // 128 A rows + 128 B rows, x2 buf

  const int n_block = blockIdx.x;
  const int m_block = blockIdx.y;
  const int tid  = threadIdx.x;
  const int wave = tid >> 5;
  const int lane = tid & 31;
  const int lo = lane & 15;
  const int hi = lane >> 4;
  const int m_sub = wave & 3;   // 4 x 32 rows  = 128
  const int n_sub = wave >> 2;  // 2 x 64 cols  = 128

  // Cooperative staging: thread t owns one 32-element row slice per k-step.
  const __bf16* grow = (tid < 128)
      ? (A + (size_t)(m_block * 128 + tid) * K)
      : (B + (size_t)(n_block * 128 + (tid - 128)) * K);
  const int lrow = tid * PAD_ROW;

  v8f acc[8] = {};

  int p = 0;
  stage_row32(grow, &lds[0][lrow]);
  stage_commit_wait();

  for (int kb = 0; kb < K; kb += BK) {
    // Prefetch next k-block into the alternate LDS buffer (overlaps WMMAs).
    if (kb + BK < K)
      stage_row32(grow + kb + BK, &lds[p ^ 1][lrow]);

    const __bf16* la = &lds[p][(m_sub * 32) * PAD_ROW];
    const __bf16* lb = &lds[p][(128 + n_sub * 64) * PAD_ROW];

    BFrag a0, a1;
    a0.h[0] = *(const v8bf*)(la + lo * PAD_ROW + 16 * hi);
    a0.h[1] = *(const v8bf*)(la + lo * PAD_ROW + 8 + 16 * hi);
    a1.h[0] = *(const v8bf*)(la + (16 + lo) * PAD_ROW + 16 * hi);
    a1.h[1] = *(const v8bf*)(la + (16 + lo) * PAD_ROW + 8 + 16 * hi);

#pragma unroll
    for (int j = 0; j < 4; ++j) {
      BFrag b;
      const __bf16* pb = lb + (j * 16 + lo) * PAD_ROW + 16 * hi;
      b.h[0] = *(const v8bf*)(pb);
      b.h[1] = *(const v8bf*)(pb + 8);
      acc[j] = __builtin_amdgcn_wmma_f32_16x16x32_bf16(
          false, a0.v, false, b.v, (short)0, acc[j], false, false);
      acc[4 + j] = __builtin_amdgcn_wmma_f32_16x16x32_bf16(
          false, a1.v, false, b.v, (short)0, acc[4 + j], false, false);
    }

    stage_commit_wait();   // next buffer resident; all reads of lds[p] done
    p ^= 1;
  }

#pragma unroll
  for (int f = 0; f < 2; ++f)
#pragma unroll
    for (int j = 0; j < 4; ++j) {
      const int n = n_block * 128 + n_sub * 64 + j * 16 + lo;
#pragma unroll
      for (int r = 0; r < 8; ++r) {
        const int m = m_block * 128 + m_sub * 32 + f * 16 + r + 8 * hi;
        C[(size_t)m * ldc + n] = acc[f * 4 + j][r];
      }
    }
}

// ===========================================================================
// Small GEMMs (scores / PV): one wave computes a 16x64 tile straight from
// global (operands are small and L2-resident).
// ===========================================================================
template<bool STORE_BF16>
__device__ __forceinline__ void gemm_wave_16x64(
    const __bf16* __restrict__ A, int lda,
    const __bf16* __restrict__ B, int ldb,
    void* __restrict__ C, int ldc,
    int m0, int n0, int K)
{
  const int lane = threadIdx.x & 31;
  const int lo = lane & 15;
  const int hi = lane >> 4;

  v8f acc[4] = {};

  const __bf16* pa  = A + (size_t)(m0 + lo) * lda;
  const __bf16* pb0 = B + (size_t)(n0 + lo) * ldb + 16 * hi;
  const int aoff = 16 * hi;

  for (int kb = 0; kb < K; kb += 32) {
    BFrag a;
    a.h[0] = *(const v8bf*)(pa + kb + aoff);
    a.h[1] = *(const v8bf*)(pa + kb + 8 + aoff);
#pragma unroll
    for (int j = 0; j < 4; ++j) {
      BFrag b;
      const __bf16* pb = pb0 + (size_t)j * 16 * ldb + kb;
      b.h[0] = *(const v8bf*)(pb);
      b.h[1] = *(const v8bf*)(pb + 8);
      acc[j] = __builtin_amdgcn_wmma_f32_16x16x32_bf16(
          false, a.v, false, b.v, (short)0, acc[j], false, false);
    }
  }

#pragma unroll
  for (int j = 0; j < 4; ++j) {
    const int n = n0 + 16 * j + lo;
#pragma unroll
    for (int r = 0; r < 8; ++r) {
      const int m = m0 + r + 8 * hi;
      if (STORE_BF16)
        ((__bf16*)C)[(size_t)m * ldc + n] = (__bf16)acc[j][r];
      else
        ((float*)C)[(size_t)m * ldc + n] = acc[j][r];
    }
  }
}

// scores[c] = Q[rep[c]] @ K[rep[c]]^T  (1024x1024, K=128), batched over clusters
__global__ void gemm_scores_kernel(const __bf16* __restrict__ qh,
                                   const __bf16* __restrict__ kh,
                                   float* __restrict__ scores,
                                   const int* __restrict__ rep)
{
  const int c = blockIdx.y;
  const int r = rep[c];
  const int wid = blockIdx.x * (blockDim.x >> 5) + (threadIdx.x >> 5); // 0..1023
  const int m_tile = wid >> 4;
  const int n_grp  = wid & 15;
  gemm_wave_16x64<false>(qh + (size_t)r * S_LEN * HD, HD,
                         kh + (size_t)r * S_LEN * HD, HD,
                         scores + (size_t)c * S_LEN * S_LEN, S_LEN,
                         m_tile * 16, n_grp * 64, HD);
}

// attn[:, h*128 : h*128+128] = P[cluster[h]] @ V[h]   (1024x128, K=1024)
__global__ void gemm_pv_kernel(const __bf16* __restrict__ p,
                               const __bf16* __restrict__ vt,
                               __bf16* __restrict__ attn,
                               const int* __restrict__ cl)
{
  const int h = blockIdx.y;
  const int c = cl[h];
  const int wid = blockIdx.x * (blockDim.x >> 5) + (threadIdx.x >> 5); // 0..127
  const int m_tile = wid >> 1;
  const int n_grp  = wid & 1;
  gemm_wave_16x64<true>(p + (size_t)c * S_LEN * S_LEN, S_LEN,
                        vt + (size_t)h * HD * S_LEN, S_LEN,
                        attn + (size_t)h * HD, DIM,
                        m_tile * 16, n_grp * 64, S_LEN);
}

// ---------------------------------------------------------------------------
// Elementwise passes
// ---------------------------------------------------------------------------
__global__ void cvt_kernel(const float* __restrict__ in, __bf16* __restrict__ out, int n4)
{
  int i = blockIdx.x * blockDim.x + threadIdx.x;
  if (i < n4) {
    const float4 f = ((const float4*)in)[i];
    v4bf o = { (__bf16)f.x, (__bf16)f.y, (__bf16)f.z, (__bf16)f.w };
    ((v4bf*)out)[i] = o;
  }
}

// RoPE on q,k (reading fused f32 QKV of shape [S][3*DIM]) -> bf16 [H][S][HD]
__global__ void rope_kernel(const float* __restrict__ qkv,
                            const float* __restrict__ fcos,
                            const float* __restrict__ fsin,
                            __bf16* __restrict__ qb, __bf16* __restrict__ kb)
{
  int tid = blockIdx.x * blockDim.x + threadIdx.x;  // 2 * 32 * 1024 * 64 = 2^22
  int i = tid & 63;
  int s = (tid >> 6) & (S_LEN - 1);
  int h = (tid >> 16) & (NH - 1);
  int t = tid >> 21;                // 0 = q, 1 = k
  const float* src = qkv + (size_t)s * (3 * DIM) + t * DIM + h * HD;
  float xr = src[2 * i];
  float xi = src[2 * i + 1];
  float c  = fcos[s * (HD / 2) + i];
  float sn = fsin[s * (HD / 2) + i];
  __bf16* dst = (t ? kb : qb) + ((size_t)h * S_LEN + s) * HD;
  dst[2 * i]     = (__bf16)(xr * c - xi * sn);
  dst[2 * i + 1] = (__bf16)(xr * sn + xi * c);
}

// V -> bf16 transposed [H][HD][S]
__global__ void vprep_kernel(const float* __restrict__ qkv, __bf16* __restrict__ vt)
{
  int tid = blockIdx.x * blockDim.x + threadIdx.x;  // 2^22
  int s = tid & (S_LEN - 1);
  int d = (tid >> 10) & (HD - 1);
  int h = tid >> 17;
  vt[((size_t)h * HD + d) * S_LEN + s] =
      (__bf16)qkv[(size_t)s * (3 * DIM) + 2 * DIM + h * HD + d];
}

// Row softmax of (scores * scale + mask), one 256-thread block per row.
__global__ void softmax_kernel(const float* __restrict__ scores,
                               const float* __restrict__ mask,
                               __bf16* __restrict__ p)
{
  const int row = blockIdx.x;           // c*1024 + q
  const int q = row & (S_LEN - 1);
  const float* src = scores + (size_t)row * S_LEN;
  const float* mr  = mask + (size_t)q * S_LEN;
  __shared__ float red[256];

  float v[4];
  float mx = -3.4e38f;
#pragma unroll
  for (int j = 0; j < 4; ++j) {
    int k = threadIdx.x + 256 * j;
    v[j] = src[k] * QK_SCALE + mr[k];
    mx = fmaxf(mx, v[j]);
  }
  red[threadIdx.x] = mx; __syncthreads();
  for (int st = 128; st > 0; st >>= 1) {
    if (threadIdx.x < st) red[threadIdx.x] = fmaxf(red[threadIdx.x], red[threadIdx.x + st]);
    __syncthreads();
  }
  mx = red[0]; __syncthreads();

  float sum = 0.f;
#pragma unroll
  for (int j = 0; j < 4; ++j) { v[j] = __expf(v[j] - mx); sum += v[j]; }
  red[threadIdx.x] = sum; __syncthreads();
  for (int st = 128; st > 0; st >>= 1) {
    if (threadIdx.x < st) red[threadIdx.x] += red[threadIdx.x + st];
    __syncthreads();
  }
  const float inv = 1.0f / red[0];
#pragma unroll
  for (int j = 0; j < 4; ++j)
    p[(size_t)row * S_LEN + threadIdx.x + 256 * j] = (__bf16)(v[j] * inv);
}

// ---------------------------------------------------------------------------
// Host orchestration (all on `stream`, graph-capture safe)
// ---------------------------------------------------------------------------
extern "C" void kernel_launch(void* const* d_in, const int* in_sizes, int n_in,
                              void* d_out, int out_size, void* d_ws, size_t ws_size,
                              hipStream_t stream) {
  const float* x    = (const float*)d_in[0];
  const float* wq   = (const float*)d_in[1];
  const float* wk   = (const float*)d_in[2];
  const float* wv   = (const float*)d_in[3];
  const float* wo   = (const float*)d_in[4];
  const float* fcos = (const float*)d_in[5];
  const float* fsin = (const float*)d_in[6];
  const float* mask = (const float*)d_in[7];
  const int*   cl   = (const int*)d_in[8];
  const int*   rep  = (const int*)d_in[9];
  float* out = (float*)d_out;
  (void)in_sizes; (void)n_in; (void)out_size; (void)ws_size;

  // Workspace carve-out (~277 MB total), 256B aligned
  char* ws = (char*)d_ws;
  size_t off = 0;
  auto alloc = [&](size_t bytes) {
    void* p = ws + off;
    off = (off + bytes + 255) & ~(size_t)255;
    return p;
  };
  __bf16* wqkv_bf = (__bf16*)alloc((size_t)3 * DIM * DIM * 2); // wq|wk|wv contiguous
  __bf16* wo_bf   = (__bf16*)alloc((size_t)DIM * DIM * 2);
  __bf16* x_bf    = (__bf16*)alloc((size_t)S_LEN * DIM * 2);
  float*  qkv_f   = (float*) alloc((size_t)S_LEN * 3 * DIM * 4);
  __bf16* q_bf    = (__bf16*)alloc((size_t)NH * S_LEN * HD * 2);
  __bf16* k_bf    = (__bf16*)alloc((size_t)NH * S_LEN * HD * 2);
  __bf16* vt_bf   = (__bf16*)alloc((size_t)NH * HD * S_LEN * 2);
  float*  scr     = (float*) alloc((size_t)NC * S_LEN * S_LEN * 4);
  __bf16* p_bf    = (__bf16*)alloc((size_t)NC * S_LEN * S_LEN * 2);
  __bf16* attn_bf = (__bf16*)alloc((size_t)S_LEN * DIM * 2);

  auto cvt = [&](const float* in, __bf16* o, size_t n) {
    int n4 = (int)(n / 4);
    cvt_kernel<<<(n4 + 255) / 256, 256, 0, stream>>>(in, o, n4);
  };
  cvt(x,  x_bf,                            (size_t)S_LEN * DIM);
  cvt(wq, wqkv_bf,                         (size_t)DIM * DIM);
  cvt(wk, wqkv_bf + (size_t)DIM * DIM,     (size_t)DIM * DIM);
  cvt(wv, wqkv_bf + (size_t)2 * DIM * DIM, (size_t)DIM * DIM);
  cvt(wo, wo_bf,                           (size_t)DIM * DIM);

  // Fused QKV projection: (1024 x 4096) @ (12288 x 4096)^T -> f32 (1024 x 12288)
  {
    dim3 g((3 * DIM) / 128, S_LEN / 128);   // (96, 8)
    gemm_block_kernel<<<g, 256, 0, stream>>>(x_bf, wqkv_bf, qkv_f, 3 * DIM, DIM);
  }

  rope_kernel<<<(2 * NH * S_LEN * (HD / 2)) / 256, 256, 0, stream>>>(
      qkv_f, fcos, fsin, q_bf, k_bf);
  vprep_kernel<<<(NH * HD * S_LEN) / 256, 256, 0, stream>>>(qkv_f, vt_bf);

  // Representative-head scores, batched over 8 clusters
  {
    dim3 g(1024 / 8, NC);
    gemm_scores_kernel<<<g, 256, 0, stream>>>(q_bf, k_bf, scr, rep);
  }

  softmax_kernel<<<NC * S_LEN, 256, 0, stream>>>(scr, mask, p_bf);

  // P @ V, batched over 32 heads (P shared per cluster)
  {
    dim3 g(128 / 8, NH);
    gemm_pv_kernel<<<g, 256, 0, stream>>>(p_bf, vt_bf, attn_bf, cl);
  }

  // Output projection: (1024 x 4096) @ (4096 x 4096)^T -> f32 d_out
  {
    dim3 g(DIM / 128, S_LEN / 128);         // (32, 8)
    gemm_block_kernel<<<g, 256, 0, stream>>>(attn_bf, wo_bf, out, DIM, DIM);
  }
}
